// Aligner_1726576857422
// MI455X (gfx1250) — compile-verified, matrix operand order checked by reading, not run on hardware
//
#include <hip/hip_runtime.h>
#include <cstdint>
#include <cstddef>

// ---------------------------------------------------------------------------
// Problem constants (from reference)
// ---------------------------------------------------------------------------
#define BB       16
#define TX       512
#define TY       2048
#define DIMIN    80
#define DIMH     512
#define ATTNCH   80
#define TEMP     0.0005f
#define F32_MIN  (-3.4028234663852886e38f)

typedef __attribute__((ext_vector_type(16))) _Float16 v16h;
typedef __attribute__((ext_vector_type(8)))  _Float16 v8h;
typedef __attribute__((ext_vector_type(8)))  float    v8f;

// ---------------------------------------------------------------------------
// Weight packing: conv k=3 (C_out, C_in, 3) f32 -> (C_out, Kp) f16, tap-major
// k = tap*Cin + c ; pad k >= 3*Cin with zero.
// ---------------------------------------------------------------------------
__global__ void pack_conv3(const float* __restrict__ w, _Float16* __restrict__ out,
                           int N, int Cin, int Kp) {
    int idx = blockIdx.x * blockDim.x + threadIdx.x;
    if (idx >= N * Kp) return;
    int n = idx / Kp, k = idx % Kp;
    float v = 0.f;
    if (k < 3 * Cin) {
        int tap = k / Cin, c = k % Cin;
        v = w[(n * Cin + c) * 3 + tap];
    }
    out[idx] = (_Float16)v;
}

// conv k=1 (C_out, K) f32 -> (C_out, Kp) f16, zero-padded
__global__ void pack_dense(const float* __restrict__ w, _Float16* __restrict__ out,
                           int N, int K, int Kp) {
    int idx = blockIdx.x * blockDim.x + threadIdx.x;
    if (idx >= N * Kp) return;
    int n = idx / Kp, k = idx % Kp;
    out[idx] = (_Float16)((k < K) ? w[n * K + k] : 0.f);
}

// ---------------------------------------------------------------------------
// im2col for key branch: x (B, TX, DIMH) f32 -> A_k (B*TX, 3*DIMH) f16
// ---------------------------------------------------------------------------
__global__ void im2col_x(const float* __restrict__ x, _Float16* __restrict__ A) {
    int idx = blockIdx.x * blockDim.x + threadIdx.x;
    const int KP = 3 * DIMH; // 1536
    if (idx >= BB * TX * KP) return;
    int m = idx / KP, k = idx % KP;
    int b = m >> 9, t = m & (TX - 1);
    int tap = k / DIMH, c = k % DIMH;
    int tt = t + tap - 1;
    float v = (tt >= 0 && tt < TX) ? x[((size_t)(b * TX + tt)) * DIMH + c] : 0.f;
    A[idx] = (_Float16)v;
}

// im2col for query branch: y (B, DIMIN, TY) f32 -> A_q (B*TY, 256) f16 (pad 240->256)
__global__ void im2col_y(const float* __restrict__ y, _Float16* __restrict__ A) {
    int idx = blockIdx.x * blockDim.x + threadIdx.x;
    const int KP = 256;
    if (idx >= BB * TY * KP) return;
    int m = idx >> 8, k = idx & 255;
    float v = 0.f;
    if (k < 3 * DIMIN) {
        int b = m >> 11, t = m & (TY - 1);
        int tap = k / DIMIN, c = k % DIMIN;
        int tt = t + tap - 1;
        if (tt >= 0 && tt < TY) v = y[((size_t)(b * DIMIN + c)) * TY + tt];
    }
    A[idx] = (_Float16)v;
}

// ---------------------------------------------------------------------------
// WMMA GEMM:  C[M x N] = act( A[M x K](f16) * W[N x K]^T(f16) + bias )
// Wave tile: 16(M) x 64(N); block = 8 waves = 128 M-rows. M % 128 == 0.
// batch mode (batch_rows_a > 0): W rows are per-batch (b = m/batch_rows_a),
// W base row = b*batch_rows_w, n limited to batch_rows_w.
// ---------------------------------------------------------------------------
__global__ __launch_bounds__(256)
void gemm_wmma(const _Float16* __restrict__ A, int lda,
               const _Float16* __restrict__ W, int ldw,
               const float* __restrict__ bias,
               void* __restrict__ Cout, int ldc,
               int M, int N, int K,
               int relu, int out_half,
               int batch_rows_a, int batch_rows_w) {
    const int wave = threadIdx.x >> 5;
    const int lane = threadIdx.x & 31;
    const int m0 = blockIdx.y * 128 + wave * 16;
    const int n0 = blockIdx.x * 64;

    int wbase = 0, nlim = N;
    if (batch_rows_a > 0) {
        int b = m0 / batch_rows_a;
        wbase = b * batch_rows_w;
        nlim = batch_rows_w;
    }

    const int ksel = (lane >> 4) * 8;       // lanes 0-15: K chunk 0/+16 ; 16-31: 8/+24
    const int arow = m0 + (lane & 15);
    const _Float16* Ap = A + (size_t)arow * lda + ksel;

    const int ncol = lane & 15;
    const _Float16* Wp[4];
    bool nvalid[4];
    #pragma unroll
    for (int j = 0; j < 4; ++j) {
        int wrow = n0 + j * 16 + ncol;
        nvalid[j] = (wrow < nlim);
        int wr = nvalid[j] ? wrow : 0;
        Wp[j] = W + (size_t)(wbase + wr) * ldw + ksel;
    }

    v8f acc[4] = {};

    for (int k0 = 0; k0 < K; k0 += 32) {
        __builtin_prefetch(Ap + k0 + 64, 0, 1);     // global_prefetch_b8 next A block
        v16h a;
        ((v8h*)&a)[0] = *(const v8h*)(Ap + k0);
        ((v8h*)&a)[1] = *(const v8h*)(Ap + k0 + 16);
        #pragma unroll
        for (int j = 0; j < 4; ++j) {
            v16h bfrag;
            ((v8h*)&bfrag)[0] = *(const v8h*)(Wp[j] + k0);
            ((v8h*)&bfrag)[1] = *(const v8h*)(Wp[j] + k0 + 16);
            acc[j] = __builtin_amdgcn_wmma_f32_16x16x32_f16(
                false, a, false, bfrag, (short)0, acc[j], false, false);
        }
    }

    const int rbase = m0 + 8 * (lane >> 4);
    #pragma unroll
    for (int j = 0; j < 4; ++j) {
        int col = n0 + j * 16 + ncol;
        if (col >= nlim) continue;
        float bv = bias ? bias[col] : 0.f;
        #pragma unroll
        for (int r = 0; r < 8; ++r) {
            int row = rbase + r;
            float v = acc[j][r] + bv;
            if (relu) v = fmaxf(v, 0.f);
            if (out_half)
                ((_Float16*)Cout)[(size_t)row * ldc + col] = (_Float16)v;
            else
                ((float*)Cout)[(size_t)row * ldc + col] = v;
        }
    }
}

// ---------------------------------------------------------------------------
// k2[b*TX+tx] = sum_c k_t[row][c]^2   (k_t rows padded to 96, pads are zero)
// ---------------------------------------------------------------------------
__global__ void k_sqsum(const _Float16* __restrict__ k_t, float* __restrict__ k2) {
    int tid = blockIdx.x * blockDim.x + threadIdx.x;
    if (tid >= BB * TX) return;
    const _Float16* r = k_t + (size_t)tid * 96;
    float s = 0.f;
    #pragma unroll 4
    for (int c = 0; c < 96; ++c) { float v = (float)r[c]; s += v * v; }
    k2[tid] = s;
}

// ---------------------------------------------------------------------------
// Fused dist2 / logp / mask / softmax. One block (512 thr) per (b, ty) row.
// Writes attn_logp (B,1,TY,TX) and alignment_soft (B,TX,TY) [transposed].
// ---------------------------------------------------------------------------
__global__ __launch_bounds__(512)
void dist_softmax(const float* __restrict__ qk,
                  const _Float16* __restrict__ q_t,
                  const float* __restrict__ k2,
                  const unsigned char* __restrict__ x_mask,
                  float* __restrict__ out_logp,
                  float* __restrict__ out_soft) {
    __shared__ float red[512];
    const int bt = blockIdx.x;           // b*TY + ty
    const int b  = bt >> 11;
    const int ty = bt & (TY - 1);
    const int tx = threadIdx.x;

    // q2 = sum_c q^2 over this row (96 padded halves, pads zero)
    float qv = 0.f;
    if (tx < 96) { float v = (float)q_t[(size_t)bt * 96 + tx]; qv = v * v; }
    red[tx] = qv; __syncthreads();
    for (int s = 256; s > 0; s >>= 1) { if (tx < s) red[tx] += red[tx + s]; __syncthreads(); }
    const float q2 = red[0]; __syncthreads();

    const float qkv = qk[(size_t)bt * TX + tx];
    const float d2  = q2 + k2[b * TX + tx] - 2.f * qkv;
    float lp = -TEMP * d2;
    if (!x_mask[b * TX + tx]) lp = F32_MIN;
    out_logp[(size_t)bt * TX + tx] = lp;

    // softmax over tx
    red[tx] = lp; __syncthreads();
    for (int s = 256; s > 0; s >>= 1) { if (tx < s) red[tx] = fmaxf(red[tx], red[tx + s]); __syncthreads(); }
    const float mx = red[0]; __syncthreads();
    const float e = __expf(lp - mx);
    red[tx] = e; __syncthreads();
    for (int s = 256; s > 0; s >>= 1) { if (tx < s) red[tx] += red[tx + s]; __syncthreads(); }
    const float attn = e / red[0];

    out_soft[((size_t)(b * TX + tx)) * TY + ty] = attn;
}

// ---------------------------------------------------------------------------
// Monotonic alignment search: forward DP + backtrace. One block per batch.
// ---------------------------------------------------------------------------
__global__ __launch_bounds__(512)
void viterbi(const float* __restrict__ soft,          // (B,TX,TY)
             const unsigned char* __restrict__ x_mask,
             const unsigned char* __restrict__ y_mask,
             unsigned char* __restrict__ dirs,        // (B,TY,TX)
             float* __restrict__ mask_out,            // (B,TX,TY), pre-zeroed
             int* __restrict__ hard_out) {            // (B,TX)
    __shared__ float v[TX];
    __shared__ int counts[TX];
    const int b = blockIdx.x;
    const int i = threadIdx.x;

    const float xm = x_mask[b * TX + i] ? 1.f : 0.f;
    v[i] = 0.f;
    counts[i] = 0;
    __syncthreads();

    const float* srow = soft + ((size_t)(b * TX + i)) * TY;
    unsigned char* drow = dirs + (size_t)b * TY * TX;

    for (int j = 0; j < TY; ++j) {
        const float ym = y_mask[b * TY + j] ? 1.f : 0.f;
        const float m  = xm * ym;
        const float val = srow[j] * m;
        const float vp = v[i];
        const float v0 = (i > 0) ? v[i - 1] : -INFINITY;
        const bool  mm = vp >= v0;
        const float vmax = mm ? vp : v0;
        const float vnew = (i <= j) ? (vmax + val) : -INFINITY;
        drow[(size_t)j * TX + i] = (m > 0.f) ? (unsigned char)(mm ? 1 : 0) : (unsigned char)1;
        __syncthreads();
        v[i] = vnew;
        __syncthreads();
    }

    if (i == 0) {
        int nvalid = 0;
        for (int t = 0; t < TX; ++t) nvalid += x_mask[b * TX + t] ? 1 : 0;
        if (!y_mask[b * TY + 0]) nvalid = 0;
        int idx = nvalid - 1;
        for (int j = TY - 1; j >= 0; --j) {
            int ic = idx < 0 ? 0 : (idx > TX - 1 ? TX - 1 : idx);
            float ym = y_mask[b * TY + j] ? 1.f : 0.f;
            float xmi = x_mask[b * TX + ic] ? 1.f : 0.f;
            float mv = xmi * ym;
            mask_out[((size_t)(b * TX + ic)) * TY + j] = mv;
            counts[ic] += (int)mv;
            int step = (int)drow[(size_t)j * TX + ic];
            idx = ic + step - 1;
        }
    }
    __syncthreads();
    hard_out[b * TX + i] = counts[i];
}

// ---------------------------------------------------------------------------
// Host-side orchestration
// ---------------------------------------------------------------------------
extern "C" void kernel_launch(void* const* d_in, const int* in_sizes, int n_in,
                              void* d_out, int out_size, void* d_ws, size_t ws_size,
                              hipStream_t stream) {
    (void)in_sizes; (void)n_in; (void)out_size; (void)ws_size;

    const float*         x      = (const float*)d_in[0];   // (B, TX, DIMH)
    const unsigned char* x_mask = (const unsigned char*)d_in[1];
    const float*         y      = (const float*)d_in[2];   // (B, DIMIN, TY)
    const unsigned char* y_mask = (const unsigned char*)d_in[3];
    const float* kw1 = (const float*)d_in[4];
    const float* kb1 = (const float*)d_in[5];
    const float* kw2 = (const float*)d_in[6];
    const float* kb2 = (const float*)d_in[7];
    const float* qw1 = (const float*)d_in[8];
    const float* qb1 = (const float*)d_in[9];
    const float* qw2 = (const float*)d_in[10];
    const float* qb2 = (const float*)d_in[11];
    const float* qw3 = (const float*)d_in[12];
    const float* qb3 = (const float*)d_in[13];

    // Output layout (flat concat): hard (B*TX int32) | soft | logp | mask
    float* outf = (float*)d_out;
    int*   out_hard = (int*)outf;
    float* out_soft = outf + BB * TX;
    float* out_logp = out_soft + (size_t)BB * TX * TY;
    float* out_mask = out_logp + (size_t)BB * TY * TX;

    // Workspace carve (region0 buffers are dead before qk is produced; alias)
    char* ws = (char*)d_ws;
    size_t off = 0;
    auto carve = [&](size_t bytes) -> void* {
        void* p = ws + off;
        off += (bytes + 255) & ~(size_t)255;
        return p;
    };
    const size_t MK = (size_t)BB * TX;   // 8192
    const size_t MQ = (size_t)BB * TY;   // 32768

    _Float16* A_k  = (_Float16*)carve(MK * 1536 * 2);
    _Float16* h1   = (_Float16*)carve(MK * 1024 * 2);
    _Float16* A_q  = (_Float16*)carve(MQ * 256 * 2);
    _Float16* h2   = (_Float16*)carve(MQ * 160 * 2);
    float*    qkbuf = (float*)ws;        // aliases region0 (69.2MB >= 67.1MB)
    _Float16* k_t  = (_Float16*)carve(MK * 96 * 2);
    _Float16* h3   = (_Float16*)carve(MQ * 96 * 2);
    _Float16* q_t  = (_Float16*)carve(MQ * 96 * 2);
    unsigned char* dirs = (unsigned char*)carve((size_t)BB * TY * TX);
    float*    k2   = (float*)carve(MK * 4);
    _Float16* W1   = (_Float16*)carve((size_t)1024 * 1536 * 2);
    _Float16* W2   = (_Float16*)carve((size_t)80 * 1024 * 2);
    _Float16* QW1  = (_Float16*)carve((size_t)160 * 256 * 2);
    _Float16* QW2  = (_Float16*)carve((size_t)80 * 160 * 2);
    _Float16* QW3  = (_Float16*)carve((size_t)80 * 96 * 2);

    // Zero padded buffers + path-mask output
    hipMemsetAsync(k_t, 0, MK * 96 * 2, stream);
    hipMemsetAsync(h3,  0, MQ * 96 * 2, stream);
    hipMemsetAsync(q_t, 0, MQ * 96 * 2, stream);
    hipMemsetAsync(out_mask, 0, (size_t)BB * TX * TY * 4, stream);

    const int TB = 256;
    auto blocks = [](size_t n, int t) { return (unsigned)((n + t - 1) / t); };

    // Weight packing
    pack_conv3<<<blocks((size_t)1024 * 1536, TB), TB, 0, stream>>>(kw1, W1, 1024, 512, 1536);
    pack_conv3<<<blocks((size_t)160 * 256,  TB), TB, 0, stream>>>(qw1, QW1, 160, 80, 256);
    pack_dense<<<blocks((size_t)80 * 1024,  TB), TB, 0, stream>>>(kw2, W2, 80, 1024, 1024);
    pack_dense<<<blocks((size_t)80 * 160,   TB), TB, 0, stream>>>(qw2, QW2, 80, 160, 160);
    pack_dense<<<blocks((size_t)80 * 96,    TB), TB, 0, stream>>>(qw3, QW3, 80, 80, 96);

    // ---- key branch ----
    im2col_x<<<blocks(MK * 1536, TB), TB, 0, stream>>>(x, A_k);
    // h1 = relu(A_k @ W1^T + kb1)         [8192 x 1024, K=1536]
    gemm_wmma<<<dim3(1024 / 64, (unsigned)(MK / 128)), TB, 0, stream>>>(
        A_k, 1536, W1, 1536, kb1, h1, 1024, (int)MK, 1024, 1536, 1, 1, 0, 0);
    // k_t = h1 @ W2^T + kb2               [8192 x 80 (pad 96), K=1024]
    gemm_wmma<<<dim3(2, (unsigned)(MK / 128)), TB, 0, stream>>>(
        h1, 1024, W2, 1024, kb2, k_t, 96, (int)MK, 80, 1024, 0, 1, 0, 0);
    k_sqsum<<<blocks(MK, TB), TB, 0, stream>>>(k_t, k2);

    // ---- query branch ----
    im2col_y<<<blocks(MQ * 256, TB), TB, 0, stream>>>(y, A_q);
    // h2 = relu(A_q @ QW1^T + qb1)        [32768 x 160, K=256]
    gemm_wmma<<<dim3(3, (unsigned)(MQ / 128)), TB, 0, stream>>>(
        A_q, 256, QW1, 256, qb1, h2, 160, (int)MQ, 160, 256, 1, 1, 0, 0);
    // h3 = relu(h2 @ QW2^T + qb2)         [32768 x 80 (pad 96), K=160]
    gemm_wmma<<<dim3(2, (unsigned)(MQ / 128)), TB, 0, stream>>>(
        h2, 160, QW2, 160, qb2, h3, 96, (int)MQ, 80, 160, 1, 1, 0, 0);
    // q_t = h3 @ QW3^T + qb3              [32768 x 80 (pad 96), K=96]
    gemm_wmma<<<dim3(2, (unsigned)(MQ / 128)), TB, 0, stream>>>(
        h3, 96, QW3, 96, qb3, q_t, 96, (int)MQ, 80, 96, 0, 1, 0, 0);

    // ---- qk = per-batch q_t @ k_t^T     [per b: 2048 x 512, K=96] ----
    gemm_wmma<<<dim3(512 / 64, (unsigned)(MQ / 128)), TB, 0, stream>>>(
        q_t, 96, k_t, 96, nullptr, qkbuf, 512, (int)MQ, 512, 96, 0, 0, TY, TX);

    // ---- dist / softmax -> attn_logp + alignment_soft ----
    dist_softmax<<<(unsigned)MQ, 512, 0, stream>>>(
        qkbuf, q_t, k2, x_mask, out_logp, out_soft);

    // ---- monotonic alignment search -> alignment_mask + alignment_hard ----
    viterbi<<<BB, TX, 0, stream>>>(out_soft, x_mask, y_mask, dirs, out_mask, out_hard);
}
